// MultiHeadPointAttention_62878321214031
// MI455X (gfx1250) — matrix operand order repaired and checked by reading, not compiled, not run on hardware
//
#include <hip/hip_runtime.h>

typedef _Float16 h16;
typedef __attribute__((ext_vector_type(16))) _Float16 v16h;
typedef __attribute__((ext_vector_type(8)))  _Float16 v8h;
typedef __attribute__((ext_vector_type(8)))  float    v8f;

#define DEVI __device__ __forceinline__

// Flip to 0 if the assembler rejects the async-to-LDS mnemonic.
#define USE_ASYNC_GATHER 1

constexpr int Bb   = 2;
constexpr int Nn   = 8192;
constexpr int Kk   = 16;
constexpr int NODES = Bb * Nn;      // 16384
constexpr int NT_NODES = 8;         // nodes per block in fused kernel
constexpr int ROWS = NT_NODES * Kk; // 128 neighbor rows per block

// ---------------------------------------------------------------------------
// WMMA helper: D = A(16x32 f16) * B(32x16 f16) + C(16x16 f32)
// ---------------------------------------------------------------------------
DEVI v8f wmma_f16(v16h a, v16h b, v8f c) {
  return __builtin_amdgcn_wmma_f32_16x16x32_f16(
      /*neg_a=*/false, a, /*neg_b=*/false, b,
      /*c_mod=*/(short)0, c, /*reuse_a=*/false, /*reuse_b=*/false);
}

// A-fragment (16x32) from row-major f16 LDS tile.
// lanes 0-15: row M=lane, K={0..7,16..23}; lanes 16-31: row M=lane-16,
// K={8..15,24..31}.  tile points at (row0, kcol0).
DEVI v16h load_a(const h16* tile, int lane, int stride) {
  const h16* p = tile + (lane & 15) * stride + ((lane & 16) ? 8 : 0);
  v8h lo = *(const v8h*)p;
  v8h hi = *(const v8h*)(p + 16);
  v16h a;
#pragma unroll
  for (int i = 0; i < 8; ++i) { a[i] = lo[i]; a[i + 8] = hi[i]; }
  return a;
}

// B-fragment: pre-packed by pack_weights, one v16h per lane per tile.
DEVI v16h load_b(const h16* pack, int tile, int lane) {
  return ((const v16h*)pack)[tile * 32 + lane];
}

#if USE_ASYNC_GATHER
// 16B per lane, global -> LDS, tracked by ASYNCcnt.
DEVI void async_ld16(void* lds, const void* g) {
  unsigned l = (unsigned)(uintptr_t)lds;            // LDS byte offset
  unsigned long long a = (unsigned long long)(uintptr_t)g;
  asm volatile("global_load_async_to_lds_b128 %0, %1, off"
               :: "v"(l), "v"(a) : "memory");
}
DEVI void wait_async0() { asm volatile("s_wait_asynccnt 0" ::: "memory"); }
#endif

// ---------------------------------------------------------------------------
// Weight packing: f32 row-major [Krows x Ndim] -> f16 fragment-ready tiles.
// tileIdx = kt*NT + nt ; element i of lane l is W[k][n] with
//   k = kt*32 + (i&7) + (l>=16 ? 8 : 0) + (i>=8 ? 16 : 0),  n = nt*16 + (l&15)
// ---------------------------------------------------------------------------
__global__ __launch_bounds__(256) void pack_weights(
    const float* Wq, const float* Wkv, const float* Wp1, const float* Wp2,
    const float* Wa1, const float* Wa2, const float* Wo,
    h16* qkv_p, h16* p1_p, h16* p2_p, h16* a1_p, h16* a2_p, h16* o_p) {
  const int EQKV = 2 * 24 * 512;  // 24576  (K=64, N=384)
  const int EP1  = 1 * 8 * 512;   // 4096   (K=32 pad, N=128)
  const int E128 = 4 * 8 * 512;   // 16384  (K=128, N=128)
  const int TOT  = EQKV + EP1 + 4 * E128;

  auto decode = [](int e, int NT, int& k, int& n) {
    int tile = e >> 9, wi = e & 511, lane = wi >> 4, i = wi & 15;
    int kt = tile / NT, nt = tile % NT;
    k = kt * 32 + (i & 7) + ((lane & 16) ? 8 : 0) + ((i & 8) ? 16 : 0);
    n = nt * 16 + (lane & 15);
  };

  for (int e = blockIdx.x * 256 + threadIdx.x; e < TOT; e += gridDim.x * 256) {
    int k, n;
    if (e < EQKV) {
      decode(e, 24, k, n);
      float v = (n < 128) ? Wq[k * 128 + n] : Wkv[k * 256 + (n - 128)];
      qkv_p[e] = (h16)v;
    } else if (e < EQKV + EP1) {
      int e2 = e - EQKV; decode(e2, 8, k, n);
      p1_p[e2] = (h16)((k < 3) ? Wp1[k * 128 + n] : 0.f);
    } else {
      int e2 = e - EQKV - EP1;
      int seg = e2 / E128, e3 = e2 % E128;
      decode(e3, 8, k, n);
      const float* src = (seg == 0) ? Wp2 : (seg == 1) ? Wa1 : (seg == 2) ? Wa2 : Wo;
      h16* dst = (seg == 0) ? p2_p : (seg == 1) ? a1_p : (seg == 2) ? a2_p : o_p;
      dst[e3] = (h16)src[k * 128 + n];
    }
  }
}

// ---------------------------------------------------------------------------
// QKV projection: [16384 x 64] @ [64 x 384] -> q,k,v (f16, 128 ch each)
// ---------------------------------------------------------------------------
__global__ __launch_bounds__(256) void qkv_kernel(
    const float* __restrict__ x, const float* __restrict__ bq,
    const float* __restrict__ bkv, const h16* __restrict__ Wqkv_p,
    h16* __restrict__ qf, h16* __restrict__ kf, h16* __restrict__ vf) {
  __shared__ h16 xs[128 * 64];
  const int tid = threadIdx.x;
  const int g0 = blockIdx.x * 128;
  for (int t = tid; t < 128 * 64; t += 256) {
    int r = t >> 6, c = t & 63;
    xs[t] = (h16)x[(size_t)(g0 + r) * 64 + c];
  }
  __syncthreads();

  const int w = tid >> 5, lane = tid & 31;
  const int mb = w * 16, rsel = (lane & 16) ? 8 : 0, nl = lane & 15;
  v16h a0 = load_a(&xs[mb * 64 + 0], lane, 64);
  v16h a1 = load_a(&xs[mb * 64 + 32], lane, 64);

  for (int nt = 0; nt < 24; ++nt) {
    v8f acc = {};
    acc = wmma_f16(a0, load_b(Wqkv_p, 0 * 24 + nt, lane), acc);
    acc = wmma_f16(a1, load_b(Wqkv_p, 1 * 24 + nt, lane), acc);
    int n = nt * 16 + nl;
#pragma unroll
    for (int r = 0; r < 8; ++r) {
      int row = mb + r + rsel;
      size_t g = (size_t)(g0 + row);
      float v = acc[r];
      if (n < 128)       qf[g * 128 + n]         = (h16)(v + bq[n]);
      else if (n < 256)  kf[g * 128 + (n - 128)] = (h16)(v + bkv[n - 128]);
      else               vf[g * 128 + (n - 256)] = (h16)(v + bkv[n - 128]);
    }
  }
}

// ---------------------------------------------------------------------------
// Fused neighbor pipeline: 8 nodes (128 neighbor rows) per block.
// ---------------------------------------------------------------------------
__global__ __launch_bounds__(256) void fused_kernel(
    const float* __restrict__ pos, const int* __restrict__ idx,
    const float* __restrict__ bp1, const float* __restrict__ bp2,
    const float* __restrict__ ba1, const float* __restrict__ ba2,
    const float* __restrict__ bo,
    const h16* __restrict__ qf, const h16* __restrict__ kf,
    const h16* __restrict__ vf,
    const h16* __restrict__ P1, const h16* __restrict__ P2,
    const h16* __restrict__ A1, const h16* __restrict__ A2,
    const h16* __restrict__ O, float* __restrict__ out) {
  extern __shared__ char smem[];
  h16* q_s   = (h16*)smem;            // 8   x128
  h16* pd_s  = q_s + 1024;            // 128 x32
  h16* k_s   = pd_s + 4096;           // 128 x128
  h16* v_s   = k_s + 16384;           // 128 x128 (becomes vpe)
  h16* act_s = v_s + 16384;           // 128 x128
  h16* rel_s = act_s + 16384;         // 128 x128 (becomes logits)
  h16* agg_s = rel_s + 16384;         // 16  x128 (rows 8..15 zero)
  int* m_s   = (int*)(agg_s + 2048);  // 128 neighbor global-row ids

  const int tid = threadIdx.x;
  const int g0 = blockIdx.x * NT_NODES;

  // phase 0a: neighbor global rows
  if (tid < ROWS) {
    int r = tid, node = g0 + (r >> 4), j = r & 15;
    int b = node / Nn, nn = node % Nn;
    m_s[r] = b * Nn + idx[(size_t)(b * Nn + nn) * Kk + j];
  }
  __syncthreads();

  // phase 0b: gather k_n / v_n rows into LDS (async direct-to-LDS path)
#if USE_ASYNC_GATHER
  for (int t = tid; t < ROWS * 16; t += 256) {  // 16 x 16B per 256B row
    int r = t >> 4, q4 = t & 15;
    size_t gm = (size_t)m_s[r];
    async_ld16(k_s + r * 128 + q4 * 8, kf + gm * 128 + q4 * 8);
    async_ld16(v_s + r * 128 + q4 * 8, vf + gm * 128 + q4 * 8);
  }
#else
  for (int t = tid; t < ROWS * 16; t += 256) {
    int r = t >> 4, q4 = t & 15;
    size_t gm = (size_t)m_s[r];
    ((uint4*)(k_s + r * 128))[q4] = ((const uint4*)(kf + gm * 128))[q4];
    ((uint4*)(v_s + r * 128))[q4] = ((const uint4*)(vf + gm * 128))[q4];
  }
#endif
  // q tile + pos-diff (padded to K=32) while the async gather is in flight
  for (int t = tid; t < 1024; t += 256) {
    int i = t >> 7, c = t & 127;
    q_s[t] = qf[(size_t)(g0 + i) * 128 + c];
  }
  for (int t = tid; t < ROWS * 32; t += 256) {
    int r = t >> 5, c = t & 31;
    int node = g0 + (r >> 4);
    float v = 0.f;
    if (c < 3) v = pos[(size_t)node * 3 + c] - pos[(size_t)m_s[r] * 3 + c];
    pd_s[t] = (h16)v;
  }
#if USE_ASYNC_GATHER
  wait_async0();
#endif
  __syncthreads();

  const int w = tid >> 5, lane = tid & 31;
  const int mb = w * 16, rsel = (lane & 16) ? 8 : 0, nl = lane & 15;

  // GEMM1: hidden = relu(pd @ Wp1 + bp1)            (1 K-step, 8 N-tiles)
  {
    v16h a = load_a(&pd_s[mb * 32], lane, 32);
    v8f acc[8] = {};
#pragma unroll
    for (int nt = 0; nt < 8; ++nt)
      acc[nt] = wmma_f16(a, load_b(P1, nt, lane), acc[nt]);
#pragma unroll
    for (int nt = 0; nt < 8; ++nt) {
      int n = nt * 16 + nl; float bias = bp1[n];
#pragma unroll
      for (int r = 0; r < 8; ++r) {
        int row = mb + r + rsel;
        float h = acc[nt][r] + bias;
        act_s[row * 128 + n] = (h16)(h > 0.f ? h : 0.f);
      }
    }
  }
  // GEMM2: pe = hidden@Wp2+bp2 ; rel = k - q + pe ; vpe = v + pe
  {
    v8f acc[8] = {};
#pragma unroll
    for (int kt = 0; kt < 4; ++kt) {
      v16h a = load_a(&act_s[mb * 128 + kt * 32], lane, 128);
#pragma unroll
      for (int nt = 0; nt < 8; ++nt)
        acc[nt] = wmma_f16(a, load_b(P2, kt * 8 + nt, lane), acc[nt]);
    }
#pragma unroll
    for (int nt = 0; nt < 8; ++nt) {
      int n = nt * 16 + nl; float bias = bp2[n];
#pragma unroll
      for (int r = 0; r < 8; ++r) {
        int row = mb + r + rsel, i = row >> 4;
        float pe  = acc[nt][r] + bias;
        float rel = (float)k_s[row * 128 + n] - (float)q_s[i * 128 + n] + pe;
        float vpe = (float)v_s[row * 128 + n] + pe;
        rel_s[row * 128 + n] = (h16)rel;
        v_s[row * 128 + n]   = (h16)vpe;
      }
    }
  }
  // GEMM3: h2 = relu(rel@Wa1+ba1)
  {
    v8f acc[8] = {};
#pragma unroll
    for (int kt = 0; kt < 4; ++kt) {
      v16h a = load_a(&rel_s[mb * 128 + kt * 32], lane, 128);
#pragma unroll
      for (int nt = 0; nt < 8; ++nt)
        acc[nt] = wmma_f16(a, load_b(A1, kt * 8 + nt, lane), acc[nt]);
    }
#pragma unroll
    for (int nt = 0; nt < 8; ++nt) {
      int n = nt * 16 + nl; float bias = ba1[n];
#pragma unroll
      for (int r = 0; r < 8; ++r) {
        int row = mb + r + rsel;
        float h = acc[nt][r] + bias;
        act_s[row * 128 + n] = (h16)(h > 0.f ? h : 0.f);
      }
    }
  }
  // GEMM4: logits = h2@Wa2+ba2  -> rel_s
  {
    v8f acc[8] = {};
#pragma unroll
    for (int kt = 0; kt < 4; ++kt) {
      v16h a = load_a(&act_s[mb * 128 + kt * 32], lane, 128);
#pragma unroll
      for (int nt = 0; nt < 8; ++nt)
        acc[nt] = wmma_f16(a, load_b(A2, kt * 8 + nt, lane), acc[nt]);
    }
#pragma unroll
    for (int nt = 0; nt < 8; ++nt) {
      int n = nt * 16 + nl; float bias = ba2[n];
#pragma unroll
      for (int r = 0; r < 8; ++r) {
        int row = mb + r + rsel;
        rel_s[row * 128 + n] = (h16)(acc[nt][r] + bias);
      }
    }
  }
  __syncthreads();

  // softmax over K=16 per (node, channel) + aggregate with vpe
  for (int t = tid; t < 1024; t += 256) agg_s[1024 + t] = (h16)0.f;  // pad rows
#pragma unroll
  for (int ii = 0; ii < 4; ++ii) {
    int col = tid + ii * 256;
    int i = col >> 7, c = col & 127;
    const h16* lp = &rel_s[i * 16 * 128 + c];
    const h16* vp = &v_s[i * 16 * 128 + c];
    float mx = -1e30f;
#pragma unroll
    for (int j = 0; j < 16; ++j) { float l = (float)lp[j * 128]; mx = l > mx ? l : mx; }
    float e[16], s = 0.f;
#pragma unroll
    for (int j = 0; j < 16; ++j) { e[j] = __expf((float)lp[j * 128] - mx); s += e[j]; }
    float inv = 1.f / s, acc = 0.f;
#pragma unroll
    for (int j = 0; j < 16; ++j) acc += e[j] * (float)vp[j * 128];
    agg_s[i * 128 + c] = (h16)(acc * inv);
  }
  __syncthreads();

  // final projection: out = agg@Wo + bo  (wave w -> N-tile w)
  {
    v8f acc = {};
#pragma unroll
    for (int kt = 0; kt < 4; ++kt)
      acc = wmma_f16(load_a(&agg_s[kt * 32], lane, 128),
                     load_b(O, kt * 8 + w, lane), acc);
    int n = w * 16 + nl; float bias = bo[n];
#pragma unroll
    for (int r = 0; r < 8; ++r) {
      int row = r + rsel;
      if (row < NT_NODES) {
        size_t node = (size_t)(g0 + row);
        out[node * 128 + n] = acc[r] + bias;
      }
    }
  }
}

// ---------------------------------------------------------------------------
extern "C" void kernel_launch(void* const* d_in, const int* in_sizes, int n_in,
                              void* d_out, int out_size, void* d_ws, size_t ws_size,
                              hipStream_t stream) {
  const float* x   = (const float*)d_in[0];
  const float* pos = (const float*)d_in[1];
  const int*   idx = (const int*)d_in[2];
  const float* Wq  = (const float*)d_in[3];  const float* bq  = (const float*)d_in[4];
  const float* Wkv = (const float*)d_in[5];  const float* bkv = (const float*)d_in[6];
  const float* Wp1 = (const float*)d_in[7];  const float* bp1 = (const float*)d_in[8];
  const float* Wp2 = (const float*)d_in[9];  const float* bp2 = (const float*)d_in[10];
  const float* Wa1 = (const float*)d_in[11]; const float* ba1 = (const float*)d_in[12];
  const float* Wa2 = (const float*)d_in[13]; const float* ba2 = (const float*)d_in[14];
  const float* Wo  = (const float*)d_in[15]; const float* bo  = (const float*)d_in[16];
  float* out = (float*)d_out;

  char* ws = (char*)d_ws;
  size_t off = 0;
  auto alloc = [&](size_t bytes) {
    char* p = ws + off;
    off = (off + bytes + 255) & ~(size_t)255;
    return p;
  };
  h16* qf   = (h16*)alloc((size_t)NODES * 128 * 2);
  h16* kf   = (h16*)alloc((size_t)NODES * 128 * 2);
  h16* vf   = (h16*)alloc((size_t)NODES * 128 * 2);
  h16* QKVp = (h16*)alloc(24576 * 2);
  h16* P1p  = (h16*)alloc(4096 * 2);
  h16* P2p  = (h16*)alloc(16384 * 2);
  h16* A1p  = (h16*)alloc(16384 * 2);
  h16* A2p  = (h16*)alloc(16384 * 2);
  h16* Op   = (h16*)alloc(16384 * 2);

  pack_weights<<<368, 256, 0, stream>>>(Wq, Wkv, Wp1, Wp2, Wa1, Wa2, Wo,
                                        QKVp, P1p, P2p, A1p, A2p, Op);
  qkv_kernel<<<NODES / 128, 256, 0, stream>>>(x, bq, bkv, QKVp, qf, kf, vf);

  const size_t smem = (1024 + 4096 + 4 * 16384 + 2048) * sizeof(h16) + 128 * sizeof(int);
  hipFuncSetAttribute((const void*)fused_kernel,
                      hipFuncAttributeMaxDynamicSharedMemorySize, (int)smem);
  fused_kernel<<<NODES / NT_NODES, 256, smem, stream>>>(
      pos, idx, bp1, bp2, ba1, ba2, bo, qf, kf, vf, P1p, P2p, A1p, A2p, Op, out);
}
